// AttentionModule_14164802142301
// MI455X (gfx1250) — matrix operand order repaired and checked by reading, not compiled, not run on hardware
//
#include <hip/hip_runtime.h>

typedef __attribute__((ext_vector_type(2)))  float  v2f;
typedef __attribute__((ext_vector_type(8)))  float  v8f;
typedef __attribute__((ext_vector_type(8)))  __bf16 v8bf;
typedef __attribute__((ext_vector_type(16))) __bf16 v16bf;

#define N_TOK 8192
#define DIM   1024
#define DQK   256
#define DV    256
#define SCALE 16.0f   // sqrt(D_QK); reference MULTIPLIES scores by it

// Scheduling-group helper: pin "loads first, then matrix ops" per region.
// mask 0x020 = VMEM read, 0x008 = MFMA/WMMA group (LLVM AMDGPU convention).
#if defined(__has_builtin)
# if __has_builtin(__builtin_amdgcn_sched_group_barrier)
#  define SCHED_GROUP(mask, size) __builtin_amdgcn_sched_group_barrier((mask), (size), 0)
# endif
#endif
#ifndef SCHED_GROUP
# define SCHED_GROUP(mask, size) do {} while (0)
#endif

// Build a bf16 WMMA A-operand chunk (16x32 slice, K-dim chunk of 32) from a
// row-major bf16 row pointer. lane L: M = L%16,
//   halves 0-7  -> k = off + j        (off already includes (L/16)*8)
//   halves 8-15 -> k = off + 16 + j
__device__ __forceinline__ v16bf load_a16(const __bf16* __restrict__ base, unsigned off) {
  v8bf lo = *(const v8bf*)(base + off);
  v8bf hi = *(const v8bf*)(base + off + 16);
  return __builtin_shufflevector(lo, hi, 0,1,2,3,4,5,6,7,8,9,10,11,12,13,14,15);
}

// ---------------------------------------------------------------------------
// Kernel 1: Q = x Wq^T, K = x Wk^T, V = x Wv^T  (full fp32 via WMMA f32 16x16x4)
// Writes Qb [N,DQK] bf16, Kb [N,DQK] bf16, Vt [DV,N] bf16 (V transposed).
// grid: 512 blocks (16-row blocks) x 256 threads (8 waves, each: 2 col-tiles x 3 mats)
// ---------------------------------------------------------------------------
__global__ __launch_bounds__(256) void qkv_proj_kernel(
    const float* __restrict__ x,  const float* __restrict__ Wq,
    const float* __restrict__ Wk, const float* __restrict__ Wv,
    __bf16* __restrict__ Qb, __bf16* __restrict__ Kb, __bf16* __restrict__ Vt)
{
  const int lane = threadIdx.x & 31;
  const int wv   = threadIdx.x >> 5;     // 0..7 -> which pair of 16-col tiles
  const int rb   = blockIdx.x;           // 0..511
  const int r16  = lane & 15;
  const int h    = lane >> 4;            // 0/1

  // 32-bit per-lane element offsets; keep SGPR base + VGPR offset addressing.
  const unsigned ao = (unsigned)(rb * 16 + r16) * DIM + h * 2;   // x A operand
  const int cb0 = wv * 32;
  unsigned wo[2];
  wo[0] = (unsigned)(cb0 + r16) * DIM + h * 2;        // W rows for col tile 0
  wo[1] = (unsigned)(cb0 + 16 + r16) * DIM + h * 2;   // ... col tile 1

  v8f zero = {};
  v8f acc[3][2];
  #pragma unroll
  for (int m = 0; m < 3; ++m)
    #pragma unroll
    for (int c = 0; c < 2; ++c) acc[m][c] = zero;

  for (int d = 0; d < DIM; d += 8) {
    // ---- batch all loads for two K-steps (k = d..d+3 and d+4..d+7) ----
    v2f a0 = *(const v2f*)(x + ao + d);
    v2f a1 = *(const v2f*)(x + ao + d + 4);
    v2f b0[3][2], b1[3][2];
    #pragma unroll
    for (int c = 0; c < 2; ++c) {
      b0[0][c] = *(const v2f*)(Wq + wo[c] + d);
      b0[1][c] = *(const v2f*)(Wk + wo[c] + d);
      b0[2][c] = *(const v2f*)(Wv + wo[c] + d);
      b1[0][c] = *(const v2f*)(Wq + wo[c] + d + 4);
      b1[1][c] = *(const v2f*)(Wk + wo[c] + d + 4);
      b1[2][c] = *(const v2f*)(Wv + wo[c] + d + 4);
    }
    // ---- then the 12 WMMAs, back-to-back ----
    #pragma unroll
    for (int m = 0; m < 3; ++m)
      #pragma unroll
      for (int c = 0; c < 2; ++c)
        acc[m][c] = __builtin_amdgcn_wmma_f32_16x16x4_f32(
            false, a0, false, b0[m][c], (short)0, acc[m][c], false, false);
    #pragma unroll
    for (int m = 0; m < 3; ++m)
      #pragma unroll
      for (int c = 0; c < 2; ++c)
        acc[m][c] = __builtin_amdgcn_wmma_f32_16x16x4_f32(
            false, a1, false, b1[m][c], (short)0, acc[m][c], false, false);

    // Pin the schedule: 14 loads up front, then 12 WMMAs back-to-back.
    SCHED_GROUP(0x020, 14);
    SCHED_GROUP(0x008, 12);
  }

  // C/D layout: lane L, vgpr r -> (row = r + h*8, col = L%16)
  #pragma unroll
  for (int c = 0; c < 2; ++c) {
    const int col = cb0 + c * 16 + r16;
    #pragma unroll
    for (int r = 0; r < 8; ++r) {
      const int row = rb * 16 + h * 8 + r;
      Qb[(size_t)row * DQK + col]   = (__bf16)acc[0][c][r];
      Kb[(size_t)row * DQK + col]   = (__bf16)acc[1][c][r];
      Vt[(size_t)col * N_TOK + row] = (__bf16)acc[2][c][r];
    }
  }
}

// ---------------------------------------------------------------------------
// Kernel 2: flash attention. One wave = one 16-row Q block x 128 output dims.
// Computes S^T tiles (A=K rows, B=Q^T) so each lane owns softmax row q=lane%16,
// and the S^T registers convert in-place into the P A-operand for P@V.
// grid: 128 blocks x 256 threads -> 1024 waves (512 row blocks x 2 segments)
// ---------------------------------------------------------------------------
__global__ __launch_bounds__(256) void flash_attn_kernel(
    const __bf16* __restrict__ Qb, const __bf16* __restrict__ Kb,
    const __bf16* __restrict__ Vt, float* __restrict__ out)
{
  const int lane = threadIdx.x & 31;
  const int wv   = threadIdx.x >> 5;
  const int gw   = blockIdx.x * 8 + wv;  // 0..1023
  const int rb   = gw >> 1;              // 16-row block
  const int seg  = gw & 1;               // which 128 of 256 out dims
  const int r16  = lane & 15;
  const int h    = lane >> 4;

  // Q as WMMA B operand (32x16 per chunk): lane L -> col q=L%16, k = 32c + h*8 + j (j<8), 32c + 16 + h*8 + (j-8) (j>=8)
  const __bf16* qrow = Qb + (size_t)(rb * 16 + r16) * DQK + h * 8;
  v16bf qB[8];
  #pragma unroll
  for (int c = 0; c < 8; ++c) qB[c] = load_a16(qrow, c * 32);

  v8f zero = {};
  v8f acc[8];
  #pragma unroll
  for (int t = 0; t < 8; ++t) acc[t] = zero;

  float m_run = -1e30f, l_run = 0.f;

  const unsigned vo = (unsigned)(seg * 128 + r16) * N_TOK + h * 8;

  for (int kv = 0; kv < N_TOK; kv += 32) {
    const unsigned ko0 = (unsigned)(kv + r16) * DQK + h * 8;        // rows kv..kv+15
    const unsigned ko1 = ko0 + 16u * DQK;                           // rows kv+16..kv+31

    // ---- S^T = K_blk @ Q^T, pipelined over 8 K-dim chunks ----
    v8f s0 = zero, s1 = zero;
    v16bf a0 = load_a16(Kb, ko0);
    v16bf a1 = load_a16(Kb, ko1);
    #pragma unroll
    for (int c = 0; c < 8; ++c) {
      v16bf na0, na1;
      if (c < 7) {
        na0 = load_a16(Kb, ko0 + (c + 1) * 32);
        na1 = load_a16(Kb, ko1 + (c + 1) * 32);
        SCHED_GROUP(0x020, 4);               // next chunk's 4 loads first...
      }
      s0 = __builtin_amdgcn_wmma_f32_16x16x32_bf16(false, a0, false, qB[c],
                                                   (short)0, s0, false, false);
      s1 = __builtin_amdgcn_wmma_f32_16x16x32_bf16(false, a1, false, qB[c],
                                                   (short)0, s1, false, false);
      SCHED_GROUP(0x008, 2);                 // ...then this chunk's 2 WMMAs
      a0 = na0; a1 = na1;
    }

    // Prefetch next kv block's K rows (L2-resident; lands in ws even at the end).
    __builtin_prefetch(Kb + ko0 + 32u * DQK, 0, 0);
    __builtin_prefetch(Kb + ko1 + 32u * DQK, 0, 0);

    // This lane owns q-row r16; it holds kv offsets h*8+r (s0) and 16+h*8+r (s1).
    float mt = -1e30f;
    #pragma unroll
    for (int r = 0; r < 8; ++r) {
      s0[r] *= SCALE; s1[r] *= SCALE;
      mt = fmaxf(mt, fmaxf(s0[r], s1[r]));
    }
    mt = fmaxf(mt, __shfl_xor(mt, 16, 32));        // combine lane pair (same q)
    const float m_new = fmaxf(m_run, mt);
    const float alpha = __expf(m_run - m_new);

    float ps = 0.f;
    v16bf pa;                                      // P as A operand, layout-free convert
    #pragma unroll
    for (int r = 0; r < 8; ++r) {
      const float p0 = __expf(s0[r] - m_new);
      const float p1 = __expf(s1[r] - m_new);
      ps += p0 + p1;
      pa[r]     = (__bf16)p0;                      // halves 0-7 : k = h*8 + r
      pa[r + 8] = (__bf16)p1;                      // halves 8-15: k = 16 + h*8 + r
    }
    ps += __shfl_xor(ps, 16, 32);
    l_run = l_run * alpha + ps;
    m_run = m_new;

    // Broadcast alpha from softmax layout (q=lane%16) to acc layout (q=r+h*8)
    float alr[8];
    #pragma unroll
    for (int r = 0; r < 8; ++r) alr[r] = __shfl(alpha, h * 8 + r, 32);

    // ---- P @ V, pipelined over 8 output tiles ----
    v16bf vb = load_a16(Vt, vo + kv);
    #pragma unroll
    for (int t = 0; t < 8; ++t) {
      v16bf nvb;
      if (t < 7) {
        nvb = load_a16(Vt, vo + kv + (unsigned)(t + 1) * 16 * N_TOK);
        SCHED_GROUP(0x020, 2);               // next tile's V loads first...
      }
      #pragma unroll
      for (int r = 0; r < 8; ++r) acc[t][r] *= alr[r];
      acc[t] = __builtin_amdgcn_wmma_f32_16x16x32_bf16(false, pa, false, vb,
                                                       (short)0, acc[t], false, false);
      SCHED_GROUP(0x008, 1);                 // ...then this tile's WMMA
      vb = nvb;
    }
  }

  float lr[8];
  #pragma unroll
  for (int r = 0; r < 8; ++r) lr[r] = 1.f / __shfl(l_run, h * 8 + r, 32);

  #pragma unroll
  for (int t = 0; t < 8; ++t) {
    const int col = seg * 128 + t * 16 + r16;
    #pragma unroll
    for (int r = 0; r < 8; ++r) {
      const int row = rb * 16 + h * 8 + r;
      out[(size_t)row * DV + col] = acc[t][r] * lr[r];
    }
  }
}

extern "C" void kernel_launch(void* const* d_in, const int* in_sizes, int n_in,
                              void* d_out, int out_size, void* d_ws, size_t ws_size,
                              hipStream_t stream) {
  (void)in_sizes; (void)n_in; (void)out_size; (void)ws_size;
  const float* x  = (const float*)d_in[0];
  const float* Wq = (const float*)d_in[1];
  const float* Wk = (const float*)d_in[2];
  const float* Wv = (const float*)d_in[3];
  float* out = (float*)d_out;

  char* ws = (char*)d_ws;                          // needs 12 MB
  __bf16* Qb = (__bf16*)(ws);
  __bf16* Kb = (__bf16*)(ws + (size_t)N_TOK * DQK * 2);
  __bf16* Vt = (__bf16*)(ws + (size_t)N_TOK * DQK * 4);

  qkv_proj_kernel<<<512, 256, 0, stream>>>(x, Wq, Wk, Wv, Qb, Kb, Vt);
  flash_attn_kernel<<<128, 256, 0, stream>>>(Qb, Kb, Vt, out);
}
